// InteractionPPBlock_4715874091792
// MI455X (gfx1250) — compile-verified, matrix operand order checked by reading, not run on hardware
//
#include <hip/hip_runtime.h>
#include <hip/hip_bf16.h>

// ---------------------------------------------------------------------------
// Problem constants (match reference)
// ---------------------------------------------------------------------------
#define E_EDGES   200000
#define HID       256
#define INTC      64
#define T_TRIP    2000000
#define SBF_DIM   42
#define NRAD      6
#define BASD      8

// GEMM tiling
#define BM 128
#define BN 128
#define BK 32
#define ASTR 40          // padded LDS row stride in bf16 elems (80B, 16B aligned)

// epilogue flags
#define F_BIAS 1
#define F_ACT  2
#define F_RBF  4
#define F_RES  8

typedef __attribute__((ext_vector_type(16))) __bf16 v16bf;
typedef __attribute__((ext_vector_type(8)))  __bf16 v8bf;
typedef __attribute__((ext_vector_type(8)))  float  v8f;

__device__ __forceinline__ __bf16 f2bf(float f) {
    unsigned u = __float_as_uint(f);
    u += 0x7FFFu + ((u >> 16) & 1u);          // round-to-nearest-even
    unsigned short h = (unsigned short)(u >> 16);
    return __builtin_bit_cast(__bf16, h);
}

__device__ __forceinline__ float silu(float x) {
    return x * (1.0f / (1.0f + __expf(-x)));
}

// CDNA5 async global->LDS copy, 16 bytes per lane, tracked by ASYNCcnt.
__device__ __forceinline__ void async_ld_b128(unsigned lds_off, const __bf16* g) {
    asm volatile("global_load_async_to_lds_b128 %0, %1, off"
                 :: "v"(lds_off), "v"(g)
                 : "memory");
}

// ---------------------------------------------------------------------------
// Generic fused bf16-WMMA GEMM:  C[M,N] = epilogue( A[M,K] @ Wt^T )
//   A:  bf16 row-major [M,K]
//   Wt: bf16 TRANSPOSED weights [N,K]  (so both LDS tiles stage contiguously)
//   epilogue: (+bias[n]) -> silu -> (* rank6 rbf modulation) -> (+res[m,n]);
//             writes fp32 C and optional bf16 mirror Cbf (feeds next GEMM's A)
// 256 threads = 8 waves; wave (wm,wn) computes 32x64 = 2x4 WMMA C tiles.
// Double-buffered async-to-LDS pipeline: 4 async B128 ops per wave per tile.
// ---------------------------------------------------------------------------
__global__ void __launch_bounds__(256)
gemm_bf16_wmma(const __bf16* __restrict__ A, const __bf16* __restrict__ Wt,
               const float* __restrict__ bias, const float* __restrict__ rbf,
               const float* __restrict__ Wrf, const float* __restrict__ res,
               float* __restrict__ C, __bf16* __restrict__ Cbf,
               int M, int N, int K, int flags)
{
    __shared__ __align__(16) __bf16 As[2][BM * ASTR];
    __shared__ __align__(16) __bf16 Bs[2][BN * ASTR];

    const int tid  = threadIdx.x;
    const int lane = tid & 31;
    const int wave = tid >> 5;
    const int wm   = wave >> 1;           // 0..3  -> rows  wm*32
    const int wn   = wave & 1;            // 0..1  -> cols  wn*64
    const int m0   = blockIdx.x * BM;
    const int n0   = blockIdx.y * BN;

    v8f acc[2][4];
#pragma unroll
    for (int i = 0; i < 2; ++i)
#pragma unroll
        for (int j = 0; j < 4; ++j)
#pragma unroll
            for (int v = 0; v < 8; ++v) acc[i][j][v] = 0.0f;

    // staging geometry: thread t copies row r = t>>1, 16-elem half cb = (t&1)*16
    const int r  = tid >> 1;
    const int cb = (tid & 1) * 16;
    int gm = m0 + r; if (gm >= M) gm = M - 1;    // clamp (no branch: keeps
    int gn = n0 + r; if (gn >= N) gn = N - 1;    //  per-wave ASYNCcnt exact)
    const __bf16* gA = A  + (size_t)gm * K + cb;
    const __bf16* gB = Wt + (size_t)gn * K + cb;
    const unsigned asb = (unsigned)(uintptr_t)&As[0][0];
    const unsigned bsb = (unsigned)(uintptr_t)&Bs[0][0];
    const unsigned abytes = (unsigned)(BM * ASTR * 2);
    const unsigned lA = asb + (unsigned)(r * ASTR + cb) * 2u;
    const unsigned lB = bsb + (unsigned)(r * ASTR + cb) * 2u;

    auto stage = [&](int buf, int kk) {
        async_ld_b128(lA + (unsigned)buf * abytes,      gA + kk);
        async_ld_b128(lA + (unsigned)buf * abytes + 16, gA + kk + 8);
        async_ld_b128(lB + (unsigned)buf * abytes,      gB + kk);
        async_ld_b128(lB + (unsigned)buf * abytes + 16, gB + kk + 8);
    };

    const int nk = K / BK;
    stage(0, 0);

    const int lrow  = lane & 15;
    const int khalf = (lane >> 4) << 3;           // 0 or 8

    for (int ki = 0; ki < nk; ++ki) {
        const int buf = ki & 1;
        if (ki + 1 < nk) {
            stage(buf ^ 1, (ki + 1) * BK);
            asm volatile("s_wait_asynccnt 0x4" ::: "memory");  // current tile done
        } else {
            asm volatile("s_wait_asynccnt 0x0" ::: "memory");
        }
        __syncthreads();

#pragma unroll
        for (int mt = 0; mt < 2; ++mt) {
            const __bf16* ap = &As[buf][(wm * 32 + mt * 16 + lrow) * ASTR];
            v8bf alo = *(const v8bf*)(ap + khalf);
            v8bf ahi = *(const v8bf*)(ap + 16 + khalf);
            v16bf a = __builtin_shufflevector(alo, ahi,
                        0,1,2,3,4,5,6,7,8,9,10,11,12,13,14,15);
#pragma unroll
            for (int nt = 0; nt < 4; ++nt) {
                const __bf16* bp = &Bs[buf][(wn * 64 + nt * 16 + lrow) * ASTR];
                v8bf blo = *(const v8bf*)(bp + khalf);
                v8bf bhi = *(const v8bf*)(bp + 16 + khalf);
                v16bf b = __builtin_shufflevector(blo, bhi,
                            0,1,2,3,4,5,6,7,8,9,10,11,12,13,14,15);
                acc[mt][nt] = __builtin_amdgcn_wmma_f32_16x16x32_bf16(
                    false, a, false, b, (short)0, acc[mt][nt], false, false);
            }
        }
        __syncthreads();
    }

    // ---- epilogue ----
    const int mof = (lane >> 4) << 3;             // lanes 16-31 hold M = v+8
#pragma unroll
    for (int mt = 0; mt < 2; ++mt) {
#pragma unroll
        for (int nt = 0; nt < 4; ++nt) {
            const int n = n0 + wn * 64 + nt * 16 + lrow;
            if (n >= N) continue;
            const float bval = (flags & F_BIAS) ? bias[n] : 0.0f;
#pragma unroll
            for (int v = 0; v < 8; ++v) {
                const int m = m0 + wm * 32 + mt * 16 + v + mof;
                if (m >= M) continue;
                float val = acc[mt][nt][v] + bval;
                if (flags & F_ACT) val = silu(val);
                if (flags & F_RBF) {
                    float rm = 0.0f;
#pragma unroll
                    for (int rr = 0; rr < NRAD; ++rr)
                        rm = fmaf(rbf[(size_t)m * NRAD + rr], Wrf[rr * N + n], rm);
                    val *= rm;
                }
                if (flags & F_RES) val += res[(size_t)m * N + n];
                C[(size_t)m * N + n] = val;
                if (Cbf) Cbf[(size_t)m * N + n] = f2bf(val);
            }
        }
    }
}

// ---------------------------------------------------------------------------
// Triplet stage: m = x_kj_down[idx_kj] * (sbf @ Wsb) ; agg[idx_ji] += m
// One wave per triplet; each lane owns channels {lane, lane+32}.
// Wsb = W_sbf1 @ W_sbf2 (42x64), L0-resident. agg / x_kj_down are L2-resident.
// ---------------------------------------------------------------------------
__global__ void __launch_bounds__(256)
triplet_kernel(const float* __restrict__ sbf, const int* __restrict__ idx_kj,
               const int* __restrict__ idx_ji, const float* __restrict__ xk,
               const float* __restrict__ Wsb, float* __restrict__ agg, int T)
{
    const int lane = threadIdx.x & 31;
    const int wave = (int)((blockIdx.x * blockDim.x + threadIdx.x) >> 5);
    const int nw   = (int)((gridDim.x * blockDim.x) >> 5);
    for (int t = wave; t < T; t += nw) {
        const int kj = idx_kj[t];
        const int ji = idx_ji[t];
        const float* srow = sbf + (size_t)t * SBF_DIM;
        float o0 = 0.0f, o1 = 0.0f;
#pragma unroll
        for (int i = 0; i < SBF_DIM; ++i) {
            const float sv = srow[i];
            o0 = fmaf(sv, Wsb[i * INTC + lane],      o0);
            o1 = fmaf(sv, Wsb[i * INTC + lane + 32], o1);
        }
        o0 *= xk[(size_t)kj * INTC + lane];
        o1 *= xk[(size_t)kj * INTC + lane + 32];
        atomicAdd(&agg[(size_t)ji * INTC + lane],      o0);
        atomicAdd(&agg[(size_t)ji * INTC + lane + 32], o1);
    }
}

// ---------------------------------------------------------------------------
// Small helpers
// ---------------------------------------------------------------------------
__global__ void f32_to_bf16_k(const float* __restrict__ s, __bf16* __restrict__ d, int n) {
    int i = blockIdx.x * blockDim.x + threadIdx.x;
    if (i < n) d[i] = f2bf(s[i]);
}

// d[n*K + k] = bf16(s[k*N + n])  — weight transpose + convert
__global__ void convT_bf16_k(const float* __restrict__ s, __bf16* __restrict__ d,
                             int K, int N) {
    int i = blockIdx.x * blockDim.x + threadIdx.x;
    if (i < K * N) {
        int k = i / N, n = i - k * N;
        d[(size_t)n * K + k] = f2bf(s[i]);
    }
}

__global__ void zero_f32(float* __restrict__ p, int n) {
    int i = blockIdx.x * blockDim.x + threadIdx.x;
    if (i < n) p[i] = 0.0f;
}

// Wrf = W_rbf1[6,8] @ W_rbf2[8,256]   (grid 6, block 256)
__global__ void fuse_rbf_kernel(const float* __restrict__ W1, const float* __restrict__ W2,
                                float* __restrict__ Wrf) {
    int i = blockIdx.x, n = threadIdx.x;
    float s = 0.0f;
#pragma unroll
    for (int b = 0; b < BASD; ++b) s = fmaf(W1[i * BASD + b], W2[b * HID + n], s);
    Wrf[i * HID + n] = s;
}

// Wsb = W_sbf1[42,8] @ W_sbf2[8,64]   (grid 42, block 64)
__global__ void fuse_sbf_kernel(const float* __restrict__ W1, const float* __restrict__ W2,
                                float* __restrict__ Wsb) {
    int i = blockIdx.x, n = threadIdx.x;
    float s = 0.0f;
#pragma unroll
    for (int b = 0; b < BASD; ++b) s = fmaf(W1[i * BASD + b], W2[b * INTC + n], s);
    Wsb[i * INTC + n] = s;
}

// ---------------------------------------------------------------------------
// Host driver
// ---------------------------------------------------------------------------
extern "C" void kernel_launch(void* const* d_in, const int* in_sizes, int n_in,
                              void* d_out, int out_size, void* d_ws, size_t ws_size,
                              hipStream_t stream) {
    const float* x      = (const float*)d_in[0];
    const float* rbf    = (const float*)d_in[1];
    const float* sbf    = (const float*)d_in[2];
    const int*   idx_kj = (const int*)  d_in[3];
    const int*   idx_ji = (const int*)  d_in[4];
    const float* W_rbf1 = (const float*)d_in[5];
    const float* W_rbf2 = (const float*)d_in[6];
    const float* W_sbf1 = (const float*)d_in[7];
    const float* W_sbf2 = (const float*)d_in[8];
    const float* W_kj   = (const float*)d_in[9];
    const float* b_kj   = (const float*)d_in[10];
    const float* W_ji   = (const float*)d_in[11];
    const float* b_ji   = (const float*)d_in[12];
    const float* W_down = (const float*)d_in[13];
    const float* W_up   = (const float*)d_in[14];
    const float* rb_W1  = (const float*)d_in[15];
    const float* rb_b1  = (const float*)d_in[16];
    const float* rb_W2  = (const float*)d_in[17];
    const float* rb_b2  = (const float*)d_in[18];
    const float* W_lin  = (const float*)d_in[19];
    const float* b_lin  = (const float*)d_in[20];
    const float* ra_W1  = (const float*)d_in[21];
    const float* ra_b1  = (const float*)d_in[22];
    const float* ra_W2  = (const float*)d_in[23];
    const float* ra_b2  = (const float*)d_in[24];

    float* out = (float*)d_out;
    char*  ws  = (char*)d_ws;

    size_t off = 0;
    auto take = [&](size_t bytes) -> size_t {
        size_t o = off;
        off = (off + bytes + 255) & ~(size_t)255;
        return o;
    };

    const size_t HH = (size_t)HID * HID;
    // transposed bf16 weights [N,K]
    __bf16* wjiT   = (__bf16*)(ws + take(HH * 2));
    __bf16* wkjT   = (__bf16*)(ws + take(HH * 2));
    __bf16* wdownT = (__bf16*)(ws + take((size_t)HID * INTC * 2));
    __bf16* wupT   = (__bf16*)(ws + take((size_t)INTC * HID * 2));
    __bf16* rbw1T  = (__bf16*)(ws + take(HH * 2));
    __bf16* rbw2T  = (__bf16*)(ws + take(HH * 2));
    __bf16* wlinT  = (__bf16*)(ws + take(HH * 2));
    __bf16* raw1T  = (__bf16*)(ws + take(2 * HH * 2));
    __bf16* raw2T  = (__bf16*)(ws + take(2 * HH * 2));
    // fused fp32 basis weights
    float* Wrf = (float*)(ws + take((size_t)NRAD * HID * 4));
    float* Wsb = (float*)(ws + take((size_t)SBF_DIM * INTC * 4));
    // big fp32 intermediates
    float* P0 = (float*)(ws + take((size_t)E_EDGES * HID * 4));
    float* P1 = (float*)(ws + take((size_t)E_EDGES * HID * 4));
    float* P2 = (float*)(ws + take((size_t)E_EDGES * HID * 4));
    float* D0 = (float*)(ws + take((size_t)E_EDGES * INTC * 4));   // x_kj_down
    float* D1 = (float*)(ws + take((size_t)E_EDGES * INTC * 4));   // agg
    // bf16 activation mirrors (GEMM A inputs)
    __bf16* xbf  = (__bf16*)(ws + take((size_t)E_EDGES * HID * 2));
    __bf16* P0bf = (__bf16*)(ws + take((size_t)E_EDGES * HID * 2));
    __bf16* P1bf = (__bf16*)(ws + take((size_t)E_EDGES * HID * 2));
    __bf16* P2bf = (__bf16*)(ws + take((size_t)E_EDGES * HID * 2));
    __bf16* d1bf = (__bf16*)(ws + take((size_t)E_EDGES * INTC * 2));
    (void)ws_size; (void)in_sizes; (void)n_in; (void)out_size;

    auto convT = [&](const float* s, __bf16* d, int K, int N) {
        convT_bf16_k<<<(K * N + 255) / 256, 256, 0, stream>>>(s, d, K, N);
    };
    convT(W_ji,   wjiT,   HID, HID);
    convT(W_kj,   wkjT,   HID, HID);
    convT(W_down, wdownT, HID, INTC);
    convT(W_up,   wupT,   INTC, HID);
    convT(rb_W1,  rbw1T,  HID, HID);
    convT(rb_W2,  rbw2T,  HID, HID);
    convT(W_lin,  wlinT,  HID, HID);
    convT(ra_W1,            raw1T,      HID, HID);
    convT(ra_W1 + HH,       raw1T + HH, HID, HID);
    convT(ra_W2,            raw2T,      HID, HID);
    convT(ra_W2 + HH,       raw2T + HH, HID, HID);

    f32_to_bf16_k<<<(E_EDGES * HID + 255) / 256, 256, 0, stream>>>(x, xbf, E_EDGES * HID);

    fuse_rbf_kernel<<<NRAD, HID, 0, stream>>>(W_rbf1, W_rbf2, Wrf);
    fuse_sbf_kernel<<<SBF_DIM, INTC, 0, stream>>>(W_sbf1, W_sbf2, Wsb);

    auto gemm = [&](const __bf16* A, const __bf16* Wt, const float* bias,
                    const float* rbfp, const float* wrfp, const float* resp,
                    float* C, __bf16* Cbf, int M, int N, int K, int flags) {
        dim3 g((M + BM - 1) / BM, (N + BN - 1) / BN);
        gemm_bf16_wmma<<<g, 256, 0, stream>>>(A, Wt, bias, rbfp, wrfp, resp,
                                              C, Cbf, M, N, K, flags);
    };

    // 1) x_ji = silu(x @ W_ji + b_ji)                              -> P0
    gemm(xbf, wjiT, b_ji, nullptr, nullptr, nullptr, P0, nullptr,
         E_EDGES, HID, HID, F_BIAS | F_ACT);
    // 2) x_kj = silu(x @ W_kj + b_kj) * (rbf @ Wrf)                -> P1 (+bf16)
    gemm(xbf, wkjT, b_kj, rbf, Wrf, nullptr, P1, P1bf,
         E_EDGES, HID, HID, F_BIAS | F_ACT | F_RBF);
    // 3) x_kj_down = silu(P1 @ W_down)                             -> D0
    gemm(P1bf, wdownT, nullptr, nullptr, nullptr, nullptr, D0, nullptr,
         E_EDGES, INTC, HID, F_ACT);

    // 4) agg = segment_sum( x_kj_down[idx_kj] * (sbf @ Wsb), idx_ji )
    zero_f32<<<(E_EDGES * INTC + 255) / 256, 256, 0, stream>>>(D1, E_EDGES * INTC);
    triplet_kernel<<<4096, 256, 0, stream>>>(sbf, idx_kj, idx_ji, D0, Wsb, D1, T_TRIP);
    f32_to_bf16_k<<<(E_EDGES * INTC + 255) / 256, 256, 0, stream>>>(D1, d1bf,
                                                                    E_EDGES * INTC);

    // 5) h = silu(agg @ W_up) + x_ji                               -> P1 (+bf16)
    gemm(d1bf, wupT, nullptr, nullptr, nullptr, P0, P1, P1bf,
         E_EDGES, HID, INTC, F_ACT | F_RES);

    // 6) residual block before skip (NB=1)
    gemm(P1bf, rbw1T, rb_b1, nullptr, nullptr, nullptr, P2, P2bf,
         E_EDGES, HID, HID, F_BIAS | F_ACT);
    gemm(P2bf, rbw2T, rb_b2, nullptr, nullptr, P1, P0, P0bf,
         E_EDGES, HID, HID, F_BIAS | F_ACT | F_RES);

    // 7) h = silu(h @ W_lin + b_lin) + x                           -> P1 (+bf16)
    gemm(P0bf, wlinT, b_lin, nullptr, nullptr, x, P1, P1bf,
         E_EDGES, HID, HID, F_BIAS | F_ACT | F_RES);

    // 8) residual blocks after skip (NA=2)
    gemm(P1bf, raw1T, ra_b1, nullptr, nullptr, nullptr, P2, P2bf,
         E_EDGES, HID, HID, F_BIAS | F_ACT);
    gemm(P2bf, raw2T, ra_b2, nullptr, nullptr, P1, P0, P0bf,
         E_EDGES, HID, HID, F_BIAS | F_ACT | F_RES);

    gemm(P0bf, raw1T + HH, ra_b1 + HID, nullptr, nullptr, nullptr, P2, P2bf,
         E_EDGES, HID, HID, F_BIAS | F_ACT);
    gemm(P2bf, raw2T + HH, ra_b2 + HID, nullptr, nullptr, P0, out, nullptr,
         E_EDGES, HID, HID, F_BIAS | F_ACT | F_RES);
}
